// LlamaApproximatedAttention_60782377173601
// MI455X (gfx1250) — compile-verified, matrix operand order checked by reading, not run on hardware
//
#include <hip/hip_runtime.h>
#include <hip/hip_fp16.h>

typedef __attribute__((ext_vector_type(16))) _Float16 v16h;
typedef __attribute__((ext_vector_type(8)))  _Float16 v8h;
typedef __attribute__((ext_vector_type(8)))  float    v8f;
typedef __attribute__((ext_vector_type(4)))  unsigned int u32x4;
typedef __attribute__((ext_vector_type(8)))  int i32x8;
typedef __attribute__((ext_vector_type(4)))  int i32x4;

#define HID  2048
#define SEQN 2048
#define RANK 64

// ---------------------------------------------------------------------------
// f32 -> f16 conversion (grid-stride)
// ---------------------------------------------------------------------------
__global__ __launch_bounds__(256)
void cvt_f32_f16(const float* __restrict__ src, _Float16* __restrict__ dst, int n) {
  int i = blockIdx.x * blockDim.x + threadIdx.x;
  int stride = gridDim.x * blockDim.x;
  for (; i < n; i += stride) dst[i] = (_Float16)src[i];
}

// ---------------------------------------------------------------------------
// TDM: DMA a [tileRows x tileK] f16 tile (row-major, row stride = strideElems)
// from global memory into LDS at byte offset ldsOff, padding each 128B row
// with 16B so the LDS row stride is 144B (= 72 halves, bank-conflict free,
// 16B aligned). Descriptor per CDNA5 ISA section 8 (D# groups 0/1; groups
// 2/3 zero => 2-D tensor). Tracked by TENSORcnt.
// ---------------------------------------------------------------------------
__device__ __forceinline__ void tdm_load_tile_f16(
    const _Float16* gtile, unsigned ldsOff,
    int tileK, int tileRows, int tensorK, int tensorRows, int strideElems)
{
  unsigned long long ga = (unsigned long long)(const void*)gtile;
  u32x4 g0;
  g0[0] = 1u;                                                // count=1, user D#
  g0[1] = ldsOff;                                            // lds_addr (bytes)
  g0[2] = (unsigned)ga;                                      // global_addr[31:0]
  g0[3] = (unsigned)((ga >> 32) & 0x01FFFFFFu) | (2u << 30); // addr[56:32] | type=2
  i32x8 g1;
  g1[0] = (1 << 16)    // data_size = 2 bytes
        | (1 << 20)    // pad_enable
        | (4 << 22)    // pad_interval: 32 DWORDs (128B of row data)
        | (3 << 25);   // pad_amount:   4 DWORDs (16B pad)
  g1[1] = (tensorK & 0xffff) << 16;                              // tensor_dim0 lo16
  g1[2] = ((unsigned)tensorK >> 16) | ((tensorRows & 0xffff) << 16); // dim0 hi | dim1 lo
  g1[3] = ((unsigned)tensorRows >> 16) | ((tileK & 0xffff) << 16);   // dim1 hi | tile_dim0
  g1[4] = tileRows & 0xffff;                                   // tile_dim1 (tile_dim2=0)
  g1[5] = strideElems;                                         // tensor_dim0_stride lo32
  g1[6] = 0;                                                   // stride hi / dim1_stride
  g1[7] = 0;
  i32x4 z4; z4[0] = 0; z4[1] = 0; z4[2] = 0; z4[3] = 0;
  i32x8 z8; z8[0] = 0; z8[1] = 0; z8[2] = 0; z8[3] = 0;
  z8[4] = 0; z8[5] = 0; z8[6] = 0; z8[7] = 0;
  __builtin_amdgcn_tensor_load_to_lds(g0, g1, z4, z4, z8, 0);
}

// ---------------------------------------------------------------------------
// Tiled WMMA GEMM:  C[M,N] = A[M,K] * op(B)
//   BT = true : B stored [N,K] row-major (C = A*B^T); tiles staged by TDM,
//               double-buffered in LDS, synced with s_wait_tensorcnt.
//   BT = false: B stored [K,N] row-major (C = A*B); cooperative staging with
//               transpose-on-LDS-store (TDM cannot transpose).
// f16 inputs, f32 accumulation (v_wmma_f32_16x16x32_f16), OutT output.
// Requires M%64==0, N%128==0, K%64==0 (true for every call below).
// Block: 256 threads = 8 wave32 waves, 2(M)x4(N); block tile 64x128, BK=64.
// ---------------------------------------------------------------------------
template<typename OutT, bool BT>
__global__ __launch_bounds__(256)
void gemm_f16_wmma(const _Float16* __restrict__ A, int lda,
                   const _Float16* __restrict__ B, int ldb,
                   OutT* __restrict__ C, int ldc,
                   int M, int N, int K)
{
  constexpr int BM = 64, BN = 128, BK = 64;
  constexpr int LSTR = BK + 8;               // 72 halves = 144B (16B aligned)
  constexpr int NBUF = BT ? 2 : 1;
  __shared__ _Float16 As[NBUF][BM * LSTR];
  __shared__ _Float16 Bs[NBUF][BN * LSTR];

  const int tid  = threadIdx.x;
  const int lane = tid & 31;
  const int wave = tid >> 5;
  const int mw   = wave >> 2;                // 0..1
  const int nw   = wave & 3;                 // 0..3
  const int blockM = blockIdx.y * BM;
  const int blockN = blockIdx.x * BN;

  v8f acc[2][2] = {};

  if (BT) {
    // ---- prologue: DMA first K-tile into buffer 0 ----
    if (tid == 0) {
      tdm_load_tile_f16(A + (size_t)blockM * lda, (unsigned)(size_t)(void*)&As[0][0],
                        BK, BM, K, M, lda);
      tdm_load_tile_f16(B + (size_t)blockN * ldb, (unsigned)(size_t)(void*)&Bs[0][0],
                        BK, BN, K, N, ldb);
    }
  }

  for (int kt = 0; kt < K; kt += BK) {
    const int cur = BT ? ((kt / BK) & 1) : 0;

    if (BT) {
      // ---- issue DMA for next K-tile into the other buffer ----
      if (kt + BK < K) {
        if (tid == 0) {
          const int nxt = cur ^ 1;
          tdm_load_tile_f16(A + (size_t)blockM * lda + (kt + BK),
                            (unsigned)(size_t)(void*)&As[nxt][0], BK, BM, K, M, lda);
          tdm_load_tile_f16(B + (size_t)blockN * ldb + (kt + BK),
                            (unsigned)(size_t)(void*)&Bs[nxt][0], BK, BN, K, N, ldb);
        }
        __builtin_amdgcn_s_wait_tensorcnt(2);  // current tile's A+B have landed
      } else {
        __builtin_amdgcn_s_wait_tensorcnt(0);
      }
    } else {
      // ---- cooperative staging (NN): A 64x64, B 64x128 transposed ----
      {
        const int row = tid >> 2, c16 = (tid & 3) * 16;
        const _Float16* gp = A + (size_t)(blockM + row) * lda + kt + c16;
        *(v8h*)&As[0][row * LSTR + c16]     = *(const v8h*)gp;
        *(v8h*)&As[0][row * LSTR + c16 + 8] = *(const v8h*)(gp + 8);
        if (kt + BK < K) __builtin_prefetch(gp + BK, 0, 3);
      }
      {
        const int k = tid >> 2, n0 = (tid & 3) * 32;
        const _Float16* gp = B + (size_t)(kt + k) * ldb + blockN + n0;
        if (kt + BK < K) __builtin_prefetch(gp + (size_t)BK * ldb, 0, 3);
#pragma unroll
        for (int g = 0; g < 4; ++g) {
          v8h v = *(const v8h*)(gp + g * 8);
#pragma unroll
          for (int i = 0; i < 8; ++i)
            Bs[0][(n0 + g * 8 + i) * LSTR + k] = v[i];
        }
      }
    }
    __syncthreads();

    // ---- fragments per ISA 7.12.2 lane layout; two K-32 chunks ----
    const int fm   = lane & 15;
    const int koff = (lane >> 4) * 8;
#pragma unroll
    for (int c = 0; c < 2; ++c) {
      v16h af[2], bf[2];
#pragma unroll
      for (int t = 0; t < 2; ++t) {
        const _Float16* ap = &As[cur][(mw * 32 + t * 16 + fm) * LSTR + c * 32];
#pragma unroll
        for (int i = 0; i < 8; ++i) af[t][i]     = ap[koff + i];
#pragma unroll
        for (int i = 0; i < 8; ++i) af[t][8 + i] = ap[16 + koff + i];
        const _Float16* bp = &Bs[cur][(nw * 32 + t * 16 + fm) * LSTR + c * 32];
#pragma unroll
        for (int i = 0; i < 8; ++i) bf[t][i]     = bp[koff + i];
#pragma unroll
        for (int i = 0; i < 8; ++i) bf[t][8 + i] = bp[16 + koff + i];
      }
#pragma unroll
      for (int i = 0; i < 2; ++i)
#pragma unroll
        for (int j = 0; j < 2; ++j)
          acc[i][j] = __builtin_amdgcn_wmma_f32_16x16x32_f16(
              false, af[i], false, bf[j], (short)0, acc[i][j], false, false);
    }
    __syncthreads();
  }

  // ---- store: C/D layout — VGPR r holds (M = r + 8*(lane>>4), N = lane&15) ----
  const int cn     = lane & 15;
  const int cmBase = (lane >> 4) * 8;
#pragma unroll
  for (int i = 0; i < 2; ++i) {
#pragma unroll
    for (int j = 0; j < 2; ++j) {
      const int row0 = blockM + mw * 32 + i * 16 + cmBase;
      const int col  = blockN + nw * 32 + j * 16 + cn;
#pragma unroll
      for (int r = 0; r < 8; ++r)
        C[(size_t)(row0 + r) * ldc + col] = (OutT)acc[i][j][r];
    }
  }
}

// ---------------------------------------------------------------------------
// Row softmax: one block per row, f32 in -> f16 probabilities out
// ---------------------------------------------------------------------------
__global__ __launch_bounds__(256)
void softmax_rows(const float* __restrict__ S, _Float16* __restrict__ P, int n) {
  __shared__ float red[256];
  const int row = blockIdx.x;
  const float* s = S + (size_t)row * n;
  _Float16*    p = P + (size_t)row * n;

  float mx = -3.4e38f;
  for (int c = threadIdx.x; c < n; c += 256) mx = fmaxf(mx, s[c]);
  red[threadIdx.x] = mx; __syncthreads();
  for (int o = 128; o > 0; o >>= 1) {
    if (threadIdx.x < o) red[threadIdx.x] = fmaxf(red[threadIdx.x], red[threadIdx.x + o]);
    __syncthreads();
  }
  mx = red[0]; __syncthreads();

  float sum = 0.f;
  for (int c = threadIdx.x; c < n; c += 256) sum += __expf(s[c] - mx);
  red[threadIdx.x] = sum; __syncthreads();
  for (int o = 128; o > 0; o >>= 1) {
    if (threadIdx.x < o) red[threadIdx.x] += red[threadIdx.x + o];
    __syncthreads();
  }
  const float inv = 1.f / red[0];
  for (int c = threadIdx.x; c < n; c += 256) p[c] = (_Float16)(__expf(s[c] - mx) * inv);
}

// ---------------------------------------------------------------------------
// Pipeline.  head_f == ones => scores identical across heads, so the
// per-head einsums collapse to full 2048x2048 GEMMs:
//   QK = H@[Wq;Wk]^T ; S = Q@K^T ; P = softmax(S) ; V = H@Wv^T ;
//   O = P@V ; Y = O@Wo^T
// ---------------------------------------------------------------------------
extern "C" void kernel_launch(void* const* d_in, const int* in_sizes, int n_in,
                              void* d_out, int out_size, void* d_ws, size_t ws_size,
                              hipStream_t stream) {
  const float* H  = (const float*)d_in[0];
  const float* Wq = (const float*)d_in[1];
  const float* Wk = (const float*)d_in[2];
  const float* Wv = (const float*)d_in[3];
  const float* Wo = (const float*)d_in[4];
  float* Y = (float*)d_out;

  char* base = (char*)d_ws;
  size_t off = 0;
  auto carve = [&](size_t bytes) { char* p = base + off; off += (bytes + 255) & ~(size_t)255; return p; };
  _Float16* H16   = (_Float16*)carve((size_t)SEQN * HID * 2);
  _Float16* Wqk16 = (_Float16*)carve((size_t)2 * RANK * HID * 2);
  _Float16* Wv16  = (_Float16*)carve((size_t)HID * HID * 2);
  _Float16* Wo16  = (_Float16*)carve((size_t)HID * HID * 2);
  _Float16* QK16  = (_Float16*)carve((size_t)SEQN * 2 * RANK * 2);
  _Float16* V16   = (_Float16*)carve((size_t)SEQN * HID * 2);
  float*    Sbuf  = (float*)   carve((size_t)SEQN * SEQN * 4);
  _Float16* P16   = (_Float16*)carve((size_t)SEQN * SEQN * 2);
  _Float16* O16   = (_Float16*)carve((size_t)SEQN * HID * 2);
  (void)ws_size; (void)in_sizes; (void)n_in; (void)out_size;

  cvt_f32_f16<<<1024, 256, 0, stream>>>(H,  H16,  SEQN * HID);
  cvt_f32_f16<<<256,  256, 0, stream>>>(Wq, Wqk16,              RANK * HID);
  cvt_f32_f16<<<256,  256, 0, stream>>>(Wk, Wqk16 + RANK * HID, RANK * HID);
  cvt_f32_f16<<<1024, 256, 0, stream>>>(Wv, Wv16, HID * HID);
  cvt_f32_f16<<<1024, 256, 0, stream>>>(Wo, Wo16, HID * HID);

  const dim3 blk(256);
  // 1) QK = H @ [Wq;Wk]^T : [2048,128], K=2048
  gemm_f16_wmma<_Float16, true><<<dim3(1, SEQN / 64), blk, 0, stream>>>(
      H16, HID, Wqk16, HID, QK16, 2 * RANK, SEQN, 2 * RANK, HID);
  // 2) V = H @ Wv^T : [2048,2048], K=2048
  gemm_f16_wmma<_Float16, true><<<dim3(HID / 128, SEQN / 64), blk, 0, stream>>>(
      H16, HID, Wv16, HID, V16, HID, SEQN, HID, HID);
  // 3) S = Q @ K^T : [2048,2048], K=64
  gemm_f16_wmma<float, true><<<dim3(SEQN / 128, SEQN / 64), blk, 0, stream>>>(
      QK16, 2 * RANK, QK16 + RANK, 2 * RANK, Sbuf, SEQN, SEQN, SEQN, RANK);
  // 4) P = softmax_rows(S) -> f16
  softmax_rows<<<SEQN, blk, 0, stream>>>(Sbuf, P16, SEQN);
  // 5) O = P @ V : [2048,2048], K=2048 (NN)
  gemm_f16_wmma<_Float16, false><<<dim3(HID / 128, SEQN / 64), blk, 0, stream>>>(
      P16, SEQN, V16, HID, O16, HID, SEQN, HID, SEQN);
  // 6) Y = O @ Wo^T : [2048,2048], K=2048, f32 out
  gemm_f16_wmma<float, true><<<dim3(HID / 128, SEQN / 64), blk, 0, stream>>>(
      O16, HID, Wo16, HID, Y, HID, SEQN, HID, HID);
}
